// DecoderWithAdaptiveAttention_17274358464567
// MI455X (gfx1250) — matrix operand order repaired and checked by reading, not compiled, not run on hardware
//
#include <hip/hip_runtime.h>

// ---------------------------------------------------------------------------
// DecoderWithAdaptiveAttention for MI455X (gfx1250, wave32, WMMA).
// All GEMMs go through v_wmma_f32_16x16x32_bf16 (f32 accumulate).
// K-loop unrolled x2 (all K used are multiples of 64): two independent
// 32-K tiles per iteration -> 16 b128 loads in flight, conversion VALU of
// tile 1 co-executes with the WMMA of tile 0.
// ---------------------------------------------------------------------------

#define BB     64      // batch
#define PIX    196     // 14*14 spatial positions
#define ENCD   2048
#define ATTD   512
#define EMBD   512
#define DECD   512
#define VOCAB  10000
#define LCAP   31
#define TSTEPS 30      // L-1

typedef __attribute__((ext_vector_type(16))) __bf16 v16bf;
typedef __attribute__((ext_vector_type(8)))  float  v8f;

union BF16Frag { unsigned int u32[8]; v16bf v; };

__device__ __forceinline__ unsigned int f2bf_bits(float f) {
  unsigned int u = __float_as_uint(f);
  u += 0x7FFFu + ((u >> 16) & 1u);        // round-to-nearest-even
  return u >> 16;
}
__device__ __forceinline__ unsigned int pk2bf(float lo, float hi) {
  return f2bf_bits(lo) | (f2bf_bits(hi) << 16);
}
__device__ __forceinline__ float sigmoidf_(float x) {
  return 1.0f / (1.0f + __expf(-x));
}

// Load one 32-K A/B fragment pair starting at column kk (branch-free).
__device__ __forceinline__ void load_frags(const float* __restrict__ ap,
                                           const float* __restrict__ bp,
                                           BF16Frag& a, BF16Frag& b)
{
  const float4 a0 = *(const float4*)(ap);
  const float4 a1 = *(const float4*)(ap + 4);
  const float4 a2 = *(const float4*)(ap + 16);
  const float4 a3 = *(const float4*)(ap + 20);
  const float4 b0 = *(const float4*)(bp);
  const float4 b1 = *(const float4*)(bp + 4);
  const float4 b2 = *(const float4*)(bp + 16);
  const float4 b3 = *(const float4*)(bp + 20);
  a.u32[0] = pk2bf(a0.x, a0.y);  a.u32[1] = pk2bf(a0.z, a0.w);
  a.u32[2] = pk2bf(a1.x, a1.y);  a.u32[3] = pk2bf(a1.z, a1.w);
  a.u32[4] = pk2bf(a2.x, a2.y);  a.u32[5] = pk2bf(a2.z, a2.w);
  a.u32[6] = pk2bf(a3.x, a3.y);  a.u32[7] = pk2bf(a3.z, a3.w);
  b.u32[0] = pk2bf(b0.x, b0.y);  b.u32[1] = pk2bf(b0.z, b0.w);
  b.u32[2] = pk2bf(b1.x, b1.y);  b.u32[3] = pk2bf(b1.z, b1.w);
  b.u32[4] = pk2bf(b2.x, b2.y);  b.u32[5] = pk2bf(b2.z, b2.w);
  b.u32[6] = pk2bf(b3.x, b3.y);  b.u32[7] = pk2bf(b3.z, b3.w);
}

// ---------------------------------------------------------------------------
// Generic GEMM:  OUT[M x N] (+)= A[M x K] * W[N x K]^T + bias[N]
// A rows may be indirected through rowmap (for encoder sort gather).
// Block: 128 threads = 4 waves; block tile 16(M) x 64(N); K multiple of 64.
// Fragment layouts (gfx1250 wave32, 16-bit operands):
//   A 16x32: lane m = L&15, elem e -> K = (e<8 ? e : e+8) + 8*(L>=16)
//   B 32x16: lane n = L&15, elem e -> K = (e<8 ? e : e+8) + 8*(L>=16)
//   C/D v8f: elem p -> (M = p + 8*(L>=16), N = L&15)
// Out-of-range rows are clamped (load real data, never poison); invalid
// outputs are dropped in the epilogue — D(m,n) depends only on A row m and
// B col n, so clamped rows never pollute valid outputs.
// ---------------------------------------------------------------------------
__global__ void gemm_wmma_kernel(const float* __restrict__ A, int lda,
                                 const int* __restrict__ rowmap,
                                 const float* __restrict__ W, int ldw,
                                 const float* __restrict__ bias,
                                 float* __restrict__ OUT, int ldo,
                                 int M, int N, int K, int accumulate)
{
  const int wave = threadIdx.x >> 5;
  const int lane = threadIdx.x & 31;
  const int g    = lane >> 4;          // half-wave group (0/1)
  const int r    = lane & 15;

  const int m0 = blockIdx.y * 16;
  const int n0 = blockIdx.x * 64 + wave * 16;

  int mlog = m0 + r;          if (mlog >= M) mlog = M - 1;
  const int arow = rowmap ? rowmap[mlog] : mlog;
  const float* __restrict__ arp = A + (long long)arow * lda + 8 * g;

  int nclamp = n0 + r;        if (nclamp >= N) nclamp = N - 1;
  const float* __restrict__ wrp = W + (long long)nclamp * ldw + 8 * g;

  v8f acc = {};

  for (int kk = 0; kk < K; kk += 64) {
    BF16Frag a0f, b0f, a1f, b1f;
    load_frags(arp + kk,      wrp + kk,      a0f, b0f);
    load_frags(arp + kk + 32, wrp + kk + 32, a1f, b1f);
    if (kk + 64 < K) {                 // pull next tiles toward the caches
      __builtin_prefetch(arp + kk + 64, 0, 1);
      __builtin_prefetch(wrp + kk + 64, 0, 1);
    }
    acc = __builtin_amdgcn_wmma_f32_16x16x32_bf16(
        false, a0f.v, false, b0f.v, (short)0, acc, false, false);
    acc = __builtin_amdgcn_wmma_f32_16x16x32_bf16(
        false, a1f.v, false, b1f.v, (short)0, acc, false, false);
  }

  const int ncol = n0 + r;
  if (ncol < N) {
    const float bv = bias ? bias[ncol] : 0.0f;
#pragma unroll
    for (int p = 0; p < 8; ++p) {
      const int mr = m0 + p + 8 * g;
      if (mr < M) {
        const long long o = (long long)mr * ldo + ncol;
        const float v = acc[p] + bv;
        OUT[o] = accumulate ? (OUT[o] + v) : v;
      }
    }
  }
}

// ---------------------------------------------------------------------------
// Setup: stable argsort of -lens, sorted caps, decode lengths (+float outputs)
// ---------------------------------------------------------------------------
__global__ void sort_setup_kernel(const int* __restrict__ caplen,
                                  const int* __restrict__ caps,
                                  int* __restrict__ sort_ind,
                                  int* __restrict__ declen,
                                  int* __restrict__ caps_sorted,
                                  float* __restrict__ out_caps,
                                  float* __restrict__ out_declen,
                                  float* __restrict__ out_sortind)
{
  __shared__ int lens[BB];
  __shared__ int sind[BB];
  const int t = threadIdx.x;
  const int li = caplen[t];
  lens[t] = li;
  __syncthreads();
  int rank = 0;
  for (int k = 0; k < BB; ++k) {
    const int lk = lens[k];
    if (lk > li || (lk == li && k < t)) ++rank;   // stable descending
  }
  sind[rank] = t;
  __syncthreads();
  const int src = sind[t];
  sort_ind[t]    = src;
  out_sortind[t] = (float)src;
  const int dl = lens[src] - 1;
  declen[t]     = dl;
  out_declen[t] = (float)dl;
  for (int j = 0; j < LCAP; ++j) {
    const int cv = caps[src * LCAP + j];
    caps_sorted[t * LCAP + j] = cv;
    out_caps[t * LCAP + j]    = (float)cv;
  }
}

__global__ void rowmap_kernel(const int* __restrict__ sind, int* __restrict__ rowmap)
{
  const int i = blockIdx.x * blockDim.x + threadIdx.x;
  if (i < BB * PIX) rowmap[i] = sind[i / PIX] * PIX + (i % PIX);
}

__global__ void mean_enc_kernel(const float* __restrict__ enc,
                                const int* __restrict__ sind,
                                float* __restrict__ mean_enc)
{
  const int i = blockIdx.x * blockDim.x + threadIdx.x;   // b*2048 + d
  if (i >= BB * ENCD) return;
  const int b = i >> 11, d = i & (ENCD - 1);
  const float* base = enc + (long long)sind[b] * PIX * ENCD + d;
  float s = 0.0f;
  for (int p = 0; p < PIX; ++p) s += base[(long long)p * ENCD];
  mean_enc[i] = s * (1.0f / (float)PIX);
}

// X[t][b][0:512] = embedding[caps_sorted[b][t]];  X[t][b][512:1024] = v_g[b]
__global__ void build_x_kernel(const float* __restrict__ emb_table,
                               const int* __restrict__ caps_sorted,
                               const float* __restrict__ v_g,
                               float* __restrict__ Xall)
{
  const long long i = (long long)blockIdx.x * blockDim.x + threadIdx.x;
  if (i >= (long long)TSTEPS * BB * 1024) return;
  const int d = (int)(i & 1023);
  const int b = (int)((i >> 10) & (BB - 1));
  const int t = (int)(i >> 16);                          // 64*1024 = 2^16
  float v;
  if (d < EMBD) v = emb_table[(long long)caps_sorted[b * LCAP + t] * EMBD + d];
  else          v = v_g[b * EMBD + (d - EMBD)];
  Xall[i] = v;
}

// LSTM pointwise: reads gates(64x2048), gpre(64x512), updates h, c, s_t
__global__ void lstm_elem_kernel(const float* __restrict__ gates,
                                 const float* __restrict__ gpre,
                                 float* __restrict__ h,
                                 float* __restrict__ c,
                                 float* __restrict__ s_t)
{
  const int i = blockIdx.x * blockDim.x + threadIdx.x;   // b*512 + d
  if (i >= BB * DECD) return;
  const int b = i >> 9, d = i & (DECD - 1);
  const float cold = c[i];
  const float gt = sigmoidf_(gpre[i]);
  s_t[i] = gt * tanhf(cold);
  const float* gb = gates + b * 4 * DECD;
  const float ig = sigmoidf_(gb[d]);
  const float fg = sigmoidf_(gb[DECD + d]);
  const float gg = tanhf    (gb[2 * DECD + d]);
  const float og = sigmoidf_(gb[3 * DECD + d]);
  const float cn = fg * cold + ig * gg;
  c[i] = cn;
  h[i] = og * tanhf(cn);
}

// att[b][p] = tanh(enc_att[b][p] + dec_att[b]) . w_full + b_full (p<196)
// att[b][196] = tanh(satt[b] + dec_att[b]) . w_full + b_full
__global__ void att_score_kernel(const float* __restrict__ enc_att,
                                 const float* __restrict__ dec_att,
                                 const float* __restrict__ satt,
                                 const float* __restrict__ wfull,
                                 const float* __restrict__ bfull,
                                 float* __restrict__ att)
{
  const int gw   = (blockIdx.x * blockDim.x + threadIdx.x) >> 5;
  const int lane = threadIdx.x & 31;
  if (gw >= BB * (PIX + 1)) return;
  const int b = gw / (PIX + 1);
  const int p = gw % (PIX + 1);
  const float* e = (p < PIX) ? (enc_att + ((long long)b * PIX + p) * ATTD)
                             : (satt + b * ATTD);
  const float* dd = dec_att + b * ATTD;
  float s = 0.0f;
  for (int k = lane; k < ATTD; k += 32) s += tanhf(e[k] + dd[k]) * wfull[k];
#pragma unroll
  for (int off = 16; off > 0; off >>= 1) s += __shfl_xor(s, off, 32);
  if (lane == 0) att[b * (PIX + 1) + p] = s + bfull[0];
}

// softmax over 197, write masked alphas for step t, build s_t * alpha[:,-1]
__global__ void softmax_alpha_kernel(const float* __restrict__ att,
                                     const int* __restrict__ declen, int t,
                                     const float* __restrict__ s_t,
                                     float* __restrict__ alpha,
                                     float* __restrict__ s_scaled,
                                     float* __restrict__ alphas_out)
{
  const int b = blockIdx.x;
  const int tid = threadIdx.x;
  __shared__ float sh[256];
  const float* ab = att + b * (PIX + 1);
  float mx = -1e30f;
  for (int p = tid; p < PIX + 1; p += 256) mx = fmaxf(mx, ab[p]);
  sh[tid] = mx; __syncthreads();
  for (int s2 = 128; s2 > 0; s2 >>= 1) { if (tid < s2) sh[tid] = fmaxf(sh[tid], sh[tid + s2]); __syncthreads(); }
  const float m = sh[0]; __syncthreads();
  float sum = 0.0f;
  for (int p = tid; p < PIX + 1; p += 256) sum += __expf(ab[p] - m);
  sh[tid] = sum; __syncthreads();
  for (int s2 = 128; s2 > 0; s2 >>= 1) { if (tid < s2) sh[tid] += sh[tid + s2]; __syncthreads(); }
  const float inv = 1.0f / sh[0];
  const bool live = (t < declen[b]);
  for (int p = tid; p < PIX + 1; p += 256) {
    const float a = __expf(ab[p] - m) * inv;
    alpha[b * (PIX + 1) + p] = a;
    if (p < PIX)
      alphas_out[((long long)b * TSTEPS + t) * PIX + p] = live ? a : 0.0f;
  }
  __syncthreads();
  const float alast = alpha[b * (PIX + 1) + PIX];
  for (int d = tid; d < DECD; d += 256) s_scaled[b * DECD + d] = s_t[b * DECD + d] * alast;
}

// awe[b][d] = sum_p enc_sorted[b][p][d] * alpha[b][p]
__global__ void awe_kernel(const float* __restrict__ enc,
                           const int* __restrict__ sind,
                           const float* __restrict__ alpha,
                           float* __restrict__ awe)
{
  const int i = blockIdx.x * blockDim.x + threadIdx.x;   // b*2048 + d
  if (i >= BB * ENCD) return;
  const int b = i >> 11, d = i & (ENCD - 1);
  const float* base = enc + (long long)sind[b] * PIX * ENCD + d;
  const float* al = alpha + b * (PIX + 1);
  float s = 0.0f;
  for (int p = 0; p < PIX; ++p) s += base[(long long)p * ENCD] * al[p];
  awe[i] = s;
}

__global__ void mask_preds_kernel(float* __restrict__ preds_t, int ldo,
                                  const int* __restrict__ declen, int t)
{
  const long long i = (long long)blockIdx.x * blockDim.x + threadIdx.x;
  if (i >= (long long)BB * VOCAB) return;
  const int b = (int)(i / VOCAB);
  const int n = (int)(i % VOCAB);
  if (t >= declen[b]) preds_t[(long long)b * ldo + n] = 0.0f;
}

// ---------------------------------------------------------------------------
extern "C" void kernel_launch(void* const* d_in, const int* in_sizes, int n_in,
                              void* d_out, int out_size, void* d_ws, size_t ws_size,
                              hipStream_t stream)
{
  (void)in_sizes; (void)n_in; (void)out_size; (void)ws_size;
  // Inputs, in setup_inputs() dict order (nested dicts in insertion order):
  const float* enc0  = (const float*)d_in[0];   // encoder_out (64,14,14,2048)
  const float* v_g   = (const float*)d_in[1];   // (64,512)
  const int*   caps  = (const int*)  d_in[2];   // (64,31)
  const int*   clen  = (const int*)  d_in[3];   // (64,1)
  const float* embT  = (const float*)d_in[4];   // embedding (10000,512)
  const float* WinH  = (const float*)d_in[5];   const float* binH = (const float*)d_in[6];
  const float* WinC  = (const float*)d_in[7];   const float* binC = (const float*)d_in[8];
  const float* Wae   = (const float*)d_in[9];   const float* bae  = (const float*)d_in[10];
  const float* Wad   = (const float*)d_in[11];  const float* bad_ = (const float*)d_in[12];
  const float* Wfc   = (const float*)d_in[13];  const float* bfc  = (const float*)d_in[14];
  const float* Wfce  = (const float*)d_in[15];  const float* bfce = (const float*)d_in[16];
  const float* Wea   = (const float*)d_in[17];  const float* bea  = (const float*)d_in[18];
  const float* Wda   = (const float*)d_in[19];  const float* bda  = (const float*)d_in[20];
  const float* Wsa   = (const float*)d_in[21];  const float* bsa  = (const float*)d_in[22];
  const float* wfull = (const float*)d_in[23];  const float* bfull= (const float*)d_in[24];
  const float* Wast  = (const float*)d_in[25];  const float* bast = (const float*)d_in[26];
  const float* Wih   = (const float*)d_in[27];  // (2048,1024)
  const float* Whh   = (const float*)d_in[28];  // (2048,512)
  const float* bih   = (const float*)d_in[29];
  const float* bhh   = (const float*)d_in[30];

  // Output layout (float, concatenated in return order)
  float* out        = (float*)d_out;
  float* out_preds  = out;                               // 64*30*10000
  float* out_caps   = out + (long long)BB * TSTEPS * VOCAB;
  float* out_declen = out_caps + BB * LCAP;
  float* out_alphas = out_declen + BB;                   // 64*30*196
  float* out_sind   = out_alphas + (long long)BB * TSTEPS * PIX;

  // Workspace carve-up (all block sizes multiples of 4 floats -> 16B aligned)
  float* ws   = (float*)d_ws;
  float* Xall = ws;  ws += (long long)TSTEPS * BB * 1024;
  float* encA = ws;  ws += (long long)BB * PIX * ATTD;   // enc_att precompute
  float* meanE= ws;  ws += BB * ENCD;
  float* h    = ws;  ws += BB * DECD;
  float* c    = ws;  ws += BB * DECD;
  float* st   = ws;  ws += BB * DECD;
  float* ssc  = ws;  ws += BB * DECD;
  float* gpre = ws;  ws += BB * DECD;
  float* datt = ws;  ws += BB * DECD;
  float* satt = ws;  ws += BB * DECD;
  float* gates= ws;  ws += BB * 4 * DECD;
  float* awe  = ws;  ws += BB * ENCD;
  float* att  = ws;  ws += BB * (PIX + 1);
  float* alph = ws;  ws += BB * (PIX + 1);
  int* sind   = (int*)ws;
  int* declen = sind + BB;
  int* capsS  = declen + BB;            // 64*31
  int* rowmap = capsS + BB * LCAP;      // 12544

  auto gemm = [&](const float* A, int lda, const int* rmap,
                  const float* W, int ldw, const float* bias,
                  float* O, int ldo, int M, int N, int K, int acc) {
    dim3 grid((N + 63) / 64, (M + 15) / 16);
    gemm_wmma_kernel<<<grid, dim3(128), 0, stream>>>(A, lda, rmap, W, ldw, bias,
                                                     O, ldo, M, N, K, acc);
  };

  // ---- setup ----
  sort_setup_kernel<<<1, BB, 0, stream>>>(clen, caps, sind, declen, capsS,
                                          out_caps, out_declen, out_sind);
  rowmap_kernel<<<(BB * PIX + 255) / 256, 256, 0, stream>>>(sind, rowmap);
  mean_enc_kernel<<<(BB * ENCD) / 256, 256, 0, stream>>>(enc0, sind, meanE);
  build_x_kernel<<<(TSTEPS * BB * 1024) / 256, 256, 0, stream>>>(embT, capsS, v_g, Xall);

  // h0 / c0
  gemm(meanE, ENCD, nullptr, WinH, ENCD, binH, h, DECD, BB, DECD, ENCD, 0);
  gemm(meanE, ENCD, nullptr, WinC, ENCD, binC, c, DECD, BB, DECD, ENCD, 0);
  // enc_att over sorted encoder rows (12544 x 512)
  gemm(enc0, ENCD, rowmap, Wea, ENCD, bea, encA, ATTD, BB * PIX, ATTD, ENCD, 0);

  // ---- time steps ----
  for (int t = 0; t < TSTEPS; ++t) {
    const float* Xt = Xall + (long long)t * BB * 1024;   // [emb_t | v_g]
    // gates = X @ W_ih^T + b_ih + h @ W_hh^T + b_hh
    gemm(Xt, 1024, nullptr, Wih, 1024, bih, gates, 4 * DECD, BB, 4 * DECD, 1024, 0);
    gemm(h,  DECD, nullptr, Whh, DECD, bhh, gates, 4 * DECD, BB, 4 * DECD, DECD, 1);
    // gpre = affine_embed(emb_t) + affine_decoder(h)
    gemm(Xt, 1024, nullptr, Wae, EMBD, bae,  gpre, DECD, BB, DECD, EMBD, 0);
    gemm(h,  DECD, nullptr, Wad, DECD, bad_, gpre, DECD, BB, DECD, DECD, 1);
    // pointwise LSTM + sentinel gate (updates h, c, st)
    lstm_elem_kernel<<<(BB * DECD) / 256, 256, 0, stream>>>(gates, gpre, h, c, st);
    // dec_att(h_new), sent_att(s_t)
    gemm(h,  DECD, nullptr, Wda, DECD, bda, datt, ATTD, BB, ATTD, DECD, 0);
    gemm(st, DECD, nullptr, Wsa, DECD, bsa, satt, ATTD, BB, ATTD, DECD, 0);
    // attention scores + softmax
    att_score_kernel<<<(BB * (PIX + 1) * 32 + 255) / 256, 256, 0, stream>>>(
        encA, datt, satt, wfull, bfull, att);
    softmax_alpha_kernel<<<BB, 256, 0, stream>>>(att, declen, t, st, alph, ssc, out_alphas);
    // awe = enc^T alpha + affine_s_t(s_t * alpha_last)
    awe_kernel<<<(BB * ENCD) / 256, 256, 0, stream>>>(enc0, sind, alph, awe);
    gemm(ssc, DECD, nullptr, Wast, DECD, bast, awe, ENCD, BB, ENCD, DECD, 1);
    // preds_t = fc(h_new) + fc_encoder(awe), written straight into d_out
    float* pt = out_preds + (long long)t * VOCAB;        // row stride TSTEPS*VOCAB
    gemm(h,   DECD, nullptr, Wfc,  DECD, bfc,  pt, TSTEPS * VOCAB, BB, VOCAB, DECD, 0);
    gemm(awe, ENCD, nullptr, Wfce, ENCD, bfce, pt, TSTEPS * VOCAB, BB, VOCAB, ENCD, 1);
    mask_preds_kernel<<<(BB * VOCAB + 255) / 256, 256, 0, stream>>>(pt, TSTEPS * VOCAB, declen, t);
  }
}
